// Decoder_22995254903420
// MI455X (gfx1250) — matrix operand order repaired and checked by reading, not compiled, not run on hardware
//
#include <hip/hip_runtime.h>
#include <math.h>

// ---------------------------------------------------------------------------
// CDNA5 (gfx1250) bf16 WMMA transformer decoder with async-to-LDS pipelines.
// ---------------------------------------------------------------------------

typedef __attribute__((ext_vector_type(16))) __bf16 v16bf;
typedef __attribute__((ext_vector_type(8)))  float  v8f;

union FragBF { v16bf v; uint4 q[2]; };

#define WMMA_BF16(a, b, c) \
    __builtin_amdgcn_wmma_f32_16x16x32_bf16(false, (a), false, (b), (short)0, (c), false, false)

// Async DMA: global -> LDS, 16B per lane, tracked by ASYNCcnt.
// VDST holds the LDS byte address (generic shared addr low 32 bits = LDS offset).
__device__ __forceinline__ void async_load_b128(const __bf16* gsrc, const __bf16* lds_dst)
{
    const unsigned           ldsoff = (unsigned)(uintptr_t)lds_dst;
    const unsigned long long gaddr  = (unsigned long long)(uintptr_t)gsrc;
    asm volatile("global_load_async_to_lds_b128 %0, %1, off"
                 :: "v"(ldsoff), "v"(gaddr) : "memory");
}
#define WAIT_ASYNC_0() asm volatile("s_wait_asynccnt 0x0" ::: "memory")
#define WAIT_ASYNC_2() asm volatile("s_wait_asynccnt 0x2" ::: "memory")
#define WAIT_ASYNC_4() asm volatile("s_wait_asynccnt 0x4" ::: "memory")

// ---------------------------------------------------------------------------
// GEMM: C[M,N] = A[M,K](bf16) * Bt[N,K](bf16, N-major) + bias; optional exact
// GELU, optional f32 residual, writes f32 and/or bf16 outputs.
// 256 threads (8 waves), 128x128 tile, K-step 32, double-buffered async LDS,
// last iteration peeled so the steady-state loop is branch-free.
// Wave layout 4(M) x 2(N); each wave 32x64 = 2x4 WMMA tiles.
// ---------------------------------------------------------------------------
__global__ __launch_bounds__(256, 2)
void gemm_bf16_kernel(const __bf16* __restrict__ A,
                      const __bf16* __restrict__ Bt,
                      const float*  __restrict__ bias,
                      const float*  __restrict__ res,
                      float*        __restrict__ Cf,
                      __bf16*       __restrict__ Cbf,
                      int M, int N, int K, int act_gelu)
{
    constexpr int LDT = 40;  // bf16 elems; 80B rows: 16B aligned, bank-spread
    __shared__ __align__(16) __bf16 As[2][128 * LDT];
    __shared__ __align__(16) __bf16 Bs[2][128 * LDT];

    const int tid   = threadIdx.x;
    const int lane  = tid & 31;
    const int wave  = tid >> 5;
    const int waveM = (wave >> 1) * 32;
    const int waveN = (wave & 1) * 64;
    const int m0 = blockIdx.y * 128;
    const int n0 = blockIdx.x * 128;
    const int lm = lane & 15;
    const int lh = lane >> 4;

    v8f acc[2][4] = {};

    const int lr = tid >> 2;        // 0..63 (+64 for second row block)
    const int lc = (tid & 3) * 8;   // K offset: 0,8,16,24

    // 4 async b128 per thread per tile
    auto issue_tile = [&](int k0, int buf) {
        const __bf16* ga = A  + (size_t)(m0 + lr) * K + k0 + lc;
        const __bf16* gb = Bt + (size_t)(n0 + lr) * K + k0 + lc;
        async_load_b128(ga,                  &As[buf][lr * LDT + lc]);
        async_load_b128(ga + (size_t)64 * K, &As[buf][(lr + 64) * LDT + lc]);
        async_load_b128(gb,                  &Bs[buf][lr * LDT + lc]);
        async_load_b128(gb + (size_t)64 * K, &Bs[buf][(lr + 64) * LDT + lc]);
    };

    auto compute_tile = [&](int buf) {
        // A frag (16x32): lane<16 -> M=lane, K={0..7,16..23}; lane>=16 -> K={8..15,24..31}
        FragBF fa[2], fb[4];
        #pragma unroll
        for (int mt = 0; mt < 2; ++mt) {
            const __bf16* p = &As[buf][(waveM + mt * 16 + lm) * LDT + lh * 8];
            fa[mt].q[0] = *(const uint4*)p;
            fa[mt].q[1] = *(const uint4*)(p + 16);
        }
        // B frag (32x16): lane<16 -> N=lane, K={0..7,16..23}; lane>=16 -> K={8..15,24..31}
        #pragma unroll
        for (int nt = 0; nt < 4; ++nt) {
            const __bf16* p = &Bs[buf][(waveN + nt * 16 + lm) * LDT + lh * 8];
            fb[nt].q[0] = *(const uint4*)p;
            fb[nt].q[1] = *(const uint4*)(p + 16);
        }
        #pragma unroll
        for (int mt = 0; mt < 2; ++mt)
            #pragma unroll
            for (int nt = 0; nt < 4; ++nt)
                acc[mt][nt] = WMMA_BF16(fa[mt].v, fb[nt].v, acc[mt][nt]);
    };

    const int nk = K >> 5;
    issue_tile(0, 0);
    // steady state: branch-free prefetch of tile i+1 while computing tile i
    for (int i = 0; i < nk - 1; ++i) {
        issue_tile((i + 1) << 5, (i + 1) & 1);
        WAIT_ASYNC_4();
        __syncthreads();
        compute_tile(i & 1);
        __syncthreads();
    }
    // peeled final tile
    WAIT_ASYNC_0();
    __syncthreads();
    compute_tile((nk - 1) & 1);

    // Epilogue. C layout: VGPR r, lane<16 -> row r / lane>=16 -> row 8+r, col = lane&15
    #pragma unroll
    for (int mt = 0; mt < 2; ++mt) {
        #pragma unroll
        for (int r = 0; r < 8; ++r) {
            const int row = m0 + waveM + mt * 16 + lh * 8 + r;
            const size_t rb = (size_t)row * N;
            #pragma unroll
            for (int nt = 0; nt < 4; ++nt) {
                const int col = n0 + waveN + nt * 16 + lm;
                float v = acc[mt][nt][r];
                if (bias)     v += bias[col];
                if (act_gelu) v = 0.5f * v * (1.0f + erff(v * 0.70710678118654752f));
                if (res)      v += res[rb + col];
                if (Cf)       Cf[rb + col] = v;
                if (Cbf)      Cbf[rb + col] = (__bf16)v;
            }
        }
    }
}

// ---------------------------------------------------------------------------
// Flash attention, DK=64, H=16.  Q:[BH,T,64] K:[BH,S,64] Vt:[BH,64,S] (bf16).
// Block = 8 waves = 128 query rows of one (b,h); wave owns 16 rows.
// K/V chunks staged cooperatively via async-to-LDS, double-buffered, last
// chunk peeled for a branch-free steady-state loop.
// ---------------------------------------------------------------------------
__global__ __launch_bounds__(256, 2)
void attn_kernel(const __bf16* __restrict__ Q,
                 const __bf16* __restrict__ Kt,
                 const __bf16* __restrict__ Vt,
                 __bf16* __restrict__ O,
                 int T, int S)
{
    constexpr int LDK = 72;  // 32(s) x 64(d) chunk, 144B rows
    constexpr int LDV = 40;  // 64(d) x 32(s) chunk, 80B rows
    __shared__ __align__(16) __bf16 Ks[2][32 * LDK];
    __shared__ __align__(16) __bf16 Vs[2][64 * LDV];
    __shared__ __align__(16) __bf16 Ps[8][16 * 40];  // per-wave P transpose slab

    const int tPer  = T >> 7;
    const int bh    = blockIdx.x / tPer;
    const int qtile = (blockIdx.x % tPer) << 7;
    const int tid   = threadIdx.x;
    const int wave  = tid >> 5;
    const int lane  = tid & 31;
    const int lm = lane & 15;
    const int lh = lane >> 4;
    const int qrow = qtile + wave * 16;

    // Q row fragments (K-dim 64 -> two 16x32 A frags), once from global
    FragBF aq[2];
    #pragma unroll
    for (int kk = 0; kk < 2; ++kk) {
        const __bf16* p = Q + ((size_t)(bh * T + qrow + lm)) * 64 + kk * 32 + lh * 8;
        aq[kk].q[0] = *(const uint4*)p;
        aq[kk].q[1] = *(const uint4*)(p + 16);
    }

    // cooperative K/V tile copy: 2 async b128 per thread per chunk
    const int kr = tid >> 3, kc = (tid & 7) * 8;  // 32x64
    const int vr = tid >> 2, vc = (tid & 3) * 8;  // 64x32
    auto issue_chunk = [&](int s0, int buf) {
        async_load_b128(Kt + ((size_t)(bh * S + s0 + kr)) * 64 + kc, &Ks[buf][kr * LDK + kc]);
        async_load_b128(Vt + ((size_t)(bh * 64 + vr)) * S + s0 + vc, &Vs[buf][vr * LDV + vc]);
    };

    v8f o[4] = {};
    float mrow[8], lrow[8], alpha[8];
    #pragma unroll
    for (int r = 0; r < 8; ++r) { mrow[r] = -1e30f; lrow[r] = 0.0f; }

    auto compute_chunk = [&](int buf) {
        // scores: two 16x16 s-tiles, accumulate over DK=64 from LDS
        v8f sc[2] = {};
        #pragma unroll
        for (int st = 0; st < 2; ++st) {
            #pragma unroll
            for (int kk = 0; kk < 2; ++kk) {
                FragBF bk;
                const __bf16* p = &Ks[buf][(st * 16 + lm) * LDK + kk * 32 + lh * 8];
                bk.q[0] = *(const uint4*)p;
                bk.q[1] = *(const uint4*)(p + 16);
                sc[st] = WMMA_BF16(aq[kk].v, bk.v, sc[st]);
            }
        }
        // online softmax over the 16x32 chunk; xor masks <16 stay in each half
        #pragma unroll
        for (int r = 0; r < 8; ++r) {
            float v0 = sc[0][r] * 0.125f;   // 1/sqrt(64)
            float v1 = sc[1][r] * 0.125f;
            float mx = fmaxf(v0, v1);
            mx = fmaxf(mx, __shfl_xor(mx, 1, 32));
            mx = fmaxf(mx, __shfl_xor(mx, 2, 32));
            mx = fmaxf(mx, __shfl_xor(mx, 4, 32));
            mx = fmaxf(mx, __shfl_xor(mx, 8, 32));
            const float mnew = fmaxf(mrow[r], mx);
            const float a  = __expf(mrow[r] - mnew);
            const float p0 = __expf(v0 - mnew);
            const float p1 = __expf(v1 - mnew);
            float ps = p0 + p1;
            ps += __shfl_xor(ps, 1, 32);
            ps += __shfl_xor(ps, 2, 32);
            ps += __shfl_xor(ps, 4, 32);
            ps += __shfl_xor(ps, 8, 32);
            lrow[r] = lrow[r] * a + ps;
            mrow[r] = mnew;
            alpha[r] = a;
            // C layout -> LDS (row = lh*8+r, col = st*16 + lm)
            Ps[wave][(lh * 8 + r) * 40 + lm]      = (__bf16)p0;
            Ps[wave][(lh * 8 + r) * 40 + 16 + lm] = (__bf16)p1;
        }
        // reload P as an A fragment (wave-private slab; DS ops in-order per wave)
        FragBF ap;
        const __bf16* pp = &Ps[wave][lm * 40 + lh * 8];
        ap.q[0] = *(const uint4*)pp;
        ap.q[1] = *(const uint4*)(pp + 16);
        // rescale O, then O += P * V (V^T from LDS)
        #pragma unroll
        for (int nt = 0; nt < 4; ++nt) {
            #pragma unroll
            for (int r = 0; r < 8; ++r) o[nt][r] *= alpha[r];
            FragBF bv;
            const __bf16* pv = &Vs[buf][(nt * 16 + lm) * LDV + lh * 8];
            bv.q[0] = *(const uint4*)pv;
            bv.q[1] = *(const uint4*)(pv + 16);
            o[nt] = WMMA_BF16(ap.v, bv.v, o[nt]);
        }
    };

    const int ns = S >> 5;
    issue_chunk(0, 0);
    for (int i = 0; i < ns - 1; ++i) {
        issue_chunk((i + 1) << 5, (i + 1) & 1);
        WAIT_ASYNC_2();
        __syncthreads();
        compute_chunk(i & 1);
        __syncthreads();
    }
    WAIT_ASYNC_0();
    __syncthreads();
    compute_chunk((ns - 1) & 1);

    // normalize + merged-heads bf16 store: O[(b*T+row)*1024 + h*64 + col]
    const int h = bh & 15, b = bh >> 4;
    #pragma unroll
    for (int r = 0; r < 8; ++r) {
        const float inv = 1.0f / lrow[r];
        const int row = qrow + lh * 8 + r;
        #pragma unroll
        for (int nt = 0; nt < 4; ++nt) {
            const int col = h * 64 + nt * 16 + lm;
            O[((size_t)(b * T + row)) * 1024 + col] = (__bf16)(o[nt][r] * inv);
        }
    }
}

// ---------------------------------------------------------------------------
// Elementwise helpers
// ---------------------------------------------------------------------------
__global__ void cvt_bf16_kernel(const float* __restrict__ src, __bf16* __restrict__ dst, int n)
{
    int i = blockIdx.x * 256 + threadIdx.x;
    if (i < n) dst[i] = (__bf16)src[i];
}

// W[K,N] f32 -> Wt[N,K] bf16 (N-major = WMMA B layout with contiguous K)
__global__ void wt_convert_kernel(const float* __restrict__ w, __bf16* __restrict__ wt, int K, int N)
{
    int idx = blockIdx.x * 256 + threadIdx.x;
    if (idx >= K * N) return;
    int k = idx / N;
    int n = idx - k * N;
    wt[(size_t)n * K + k] = (__bf16)w[idx];
}

// src[b, t, colOff + h*64 + d] (f32, stride srcCols) -> dst[(b*16+h), t, d] (bf16)
__global__ void pack_heads_kernel(const float* __restrict__ src, int srcCols, int colOff,
                                  __bf16* __restrict__ dst, int rowsPerB)
{
    int idx = blockIdx.x * 256 + threadIdx.x;
    int total = 4 * rowsPerB * 1024;
    if (idx >= total) return;
    int c = idx & 1023;
    int t = (idx >> 10) % rowsPerB;
    int b = idx / (rowsPerB << 10);
    int h = c >> 6, d = c & 63;
    dst[(((size_t)(b * 16 + h)) * rowsPerB + t) * 64 + d] =
        (__bf16)src[(size_t)(b * rowsPerB + t) * srcCols + colOff + c];
}

// same but transposed per head: dst[(b*16+h), d, t]  (for V^T)
__global__ void pack_heads_t_kernel(const float* __restrict__ src, int srcCols, int colOff,
                                    __bf16* __restrict__ dst, int rowsPerB)
{
    int idx = blockIdx.x * 256 + threadIdx.x;
    int total = 4 * rowsPerB * 1024;
    if (idx >= total) return;
    int c = idx & 1023;
    int t = (idx >> 10) % rowsPerB;
    int b = idx / (rowsPerB << 10);
    int h = c >> 6, d = c & 63;
    dst[(((size_t)(b * 16 + h)) * 64 + d) * rowsPerB + t] =
        (__bf16)src[(size_t)(b * rowsPerB + t) * srcCols + colOff + c];
}

// ---------------------------------------------------------------------------
// Orchestration
// ---------------------------------------------------------------------------
extern "C" void kernel_launch(void* const* d_in, const int* in_sizes, int n_in,
                              void* d_out, int out_size, void* d_ws, size_t ws_size,
                              hipStream_t stream)
{
    constexpr int Dm = 1024, HID = 4096, Lnum = 4, Mrows = 4096, T = 1024, S = 1024;

    const float* tgt      = (const float*)d_in[0];
    const float* enc      = (const float*)d_in[1];
    const float* sa_qkv_w = (const float*)d_in[2];
    const float* sa_qkv_b = (const float*)d_in[3];
    const float* sa_out_w = (const float*)d_in[4];
    const float* sa_out_b = (const float*)d_in[5];
    const float* ca_q_w   = (const float*)d_in[6];
    const float* ca_q_b   = (const float*)d_in[7];
    const float* ca_kv_w  = (const float*)d_in[8];
    const float* ca_kv_b  = (const float*)d_in[9];
    const float* ca_out_w = (const float*)d_in[10];
    const float* ca_out_b = (const float*)d_in[11];
    const float* ff_w1    = (const float*)d_in[12];
    const float* ff_b1    = (const float*)d_in[13];
    const float* ff_w2    = (const float*)d_in[14];
    const float* ff_b2    = (const float*)d_in[15];

    char* ws = (char*)d_ws;
    size_t off = 0;
    auto alloc = [&](size_t bytes) -> void* {
        void* p = ws + off;
        off += (bytes + 255) & ~(size_t)255;
        return p;
    };

    __bf16* wbuf   = (__bf16*)alloc((size_t)HID * Dm * 2);        // max weight (4M bf16)
    float*  qkvbuf = (float*) alloc((size_t)Mrows * 3 * Dm * 4);  // qkv / q / kv scratch
    __bf16* Qbf    = (__bf16*)alloc((size_t)64 * T * 64 * 2);
    __bf16* Kbf    = (__bf16*)alloc((size_t)64 * S * 64 * 2);
    __bf16* Vtbf   = (__bf16*)alloc((size_t)64 * 64 * S * 2);
    __bf16* obf    = (__bf16*)alloc((size_t)Mrows * Dm * 2);
    __bf16* hbf    = (__bf16*)alloc((size_t)Mrows * HID * 2);
    float*  f0     = (float*) alloc((size_t)Mrows * Dm * 4);
    float*  f1     = (float*) alloc((size_t)Mrows * Dm * 4);
    float*  f2     = (float*) alloc((size_t)Mrows * Dm * 4);
    __bf16* g0     = (__bf16*)alloc((size_t)Mrows * Dm * 2);
    __bf16* g1     = (__bf16*)alloc((size_t)Mrows * Dm * 2);
    __bf16* gC     = (__bf16*)alloc((size_t)Mrows * Dm * 2);
    __bf16* encbf  = (__bf16*)alloc((size_t)Mrows * Dm * 2);

    auto ewgrid = [](int n) { return dim3((unsigned)((n + 255) / 256)); };

    auto gemm = [&](const __bf16* Aq, const float* Wsrc, const float* bias,
                    const float* resid, float* Cf, __bf16* Cbf,
                    int M, int N, int K, int gelu) {
        wt_convert_kernel<<<ewgrid(K * N), 256, 0, stream>>>(Wsrc, wbuf, K, N);
        dim3 grid(N / 128, M / 128);
        gemm_bf16_kernel<<<grid, 256, 0, stream>>>(Aq, wbuf, bias, resid, Cf, Cbf, M, N, K, gelu);
    };

    const int attnBlocks = 64 * (T / 128);  // B*H * query tiles

    cvt_bf16_kernel<<<ewgrid(Mrows * Dm), 256, 0, stream>>>(enc, encbf, Mrows * Dm);
    cvt_bf16_kernel<<<ewgrid(Mrows * Dm), 256, 0, stream>>>(tgt, gC,    Mrows * Dm);

    const float* xcur    = tgt;
    __bf16*      xcur_bf = gC;

    for (int i = 0; i < Lnum; ++i) {
        // ---- self attention ----
        gemm(xcur_bf, sa_qkv_w + (size_t)i * Dm * 3 * Dm, sa_qkv_b + (size_t)i * 3 * Dm,
             nullptr, qkvbuf, nullptr, Mrows, 3 * Dm, Dm, 0);
        pack_heads_kernel  <<<ewgrid(Mrows * Dm), 256, 0, stream>>>(qkvbuf, 3 * Dm, 0,      Qbf,  T);
        pack_heads_kernel  <<<ewgrid(Mrows * Dm), 256, 0, stream>>>(qkvbuf, 3 * Dm, Dm,     Kbf,  S);
        pack_heads_t_kernel<<<ewgrid(Mrows * Dm), 256, 0, stream>>>(qkvbuf, 3 * Dm, 2 * Dm, Vtbf, S);
        attn_kernel<<<attnBlocks, 256, 0, stream>>>(Qbf, Kbf, Vtbf, obf, T, S);
        gemm(obf, sa_out_w + (size_t)i * Dm * Dm, sa_out_b + (size_t)i * Dm,
             xcur, f0, g0, Mrows, Dm, Dm, 0);

        // ---- cross attention ----
        gemm(g0, ca_q_w + (size_t)i * Dm * Dm, ca_q_b + (size_t)i * Dm,
             nullptr, qkvbuf, nullptr, Mrows, Dm, Dm, 0);
        pack_heads_kernel<<<ewgrid(Mrows * Dm), 256, 0, stream>>>(qkvbuf, Dm, 0, Qbf, T);
        gemm(encbf, ca_kv_w + (size_t)i * Dm * 2 * Dm, ca_kv_b + (size_t)i * 2 * Dm,
             nullptr, qkvbuf, nullptr, Mrows, 2 * Dm, Dm, 0);
        pack_heads_kernel  <<<ewgrid(Mrows * Dm), 256, 0, stream>>>(qkvbuf, 2 * Dm, 0,  Kbf,  S);
        pack_heads_t_kernel<<<ewgrid(Mrows * Dm), 256, 0, stream>>>(qkvbuf, 2 * Dm, Dm, Vtbf, S);
        attn_kernel<<<attnBlocks, 256, 0, stream>>>(Qbf, Kbf, Vtbf, obf, T, S);
        gemm(obf, ca_out_w + (size_t)i * Dm * Dm, ca_out_b + (size_t)i * Dm,
             f0, f1, g1, Mrows, Dm, Dm, 0);

        // ---- feed forward ----
        gemm(g1, ff_w1 + (size_t)i * Dm * HID, ff_b1 + (size_t)i * HID,
             nullptr, nullptr, hbf, Mrows, HID, Dm, 1);
        float*  xo  = (i == Lnum - 1) ? (float*)d_out : f2;
        __bf16* xob = (i == Lnum - 1) ? nullptr : gC;
        gemm(hbf, ff_w2 + (size_t)i * HID * Dm, ff_b2 + (size_t)i * Dm,
             f1, xo, xob, Mrows, Dm, HID, 0);

        xcur = f2;
        xcur_bf = gC;
    }
}
